// BMM_S8T_S8N_BF16T_72292889526617
// MI455X (gfx1250) — 8-bit matrix operand order repaired and checked by reading, compiled, not run on hardware
//
#include <hip/hip_runtime.h>
#include <hip/hip_bf16.h>
#include <stdint.h>

typedef int v8i __attribute__((ext_vector_type(8)));
typedef int v4i __attribute__((vector_size(16)));
typedef __attribute__((address_space(1))) v4i gv4i;
typedef __attribute__((address_space(3))) v4i sv4i;

#define BATCH 32
#define MDIM  512
#define NDIM  512
#define KDIM  2048

#define BM 128
#define BN 128
#define BK 64
#define KSTEPS (KDIM / BK)
#define LDS_STRIDE 80

#if __has_builtin(__builtin_amdgcn_global_load_async_to_lds_b128)
#define HAVE_ASYNC_LDS 1
#else
#define HAVE_ASYNC_LDS 0
#endif

#if __has_builtin(__builtin_amdgcn_s_wait_asynccnt)
#define WAIT_ASYNC0() __builtin_amdgcn_s_wait_asynccnt(0)
#else
#define WAIT_ASYNC0() asm volatile("s_wait_asynccnt 0x0" ::: "memory")
#endif

#if HAVE_ASYNC_LDS
__device__ __forceinline__ void async_copy16(const int8_t* g, int8_t* l) {
    __builtin_amdgcn_global_load_async_to_lds_b128(
        (gv4i*)g, (sv4i*)l,  0,  0);
}
#endif

__global__ __launch_bounds__(256)
void bmm_s8s8_bf16_wmma(const int8_t* __restrict__ A,
                        const int8_t* __restrict__ Bm,
                        __hip_bfloat16* __restrict__ out,
                        const float* __restrict__ alphap)
{
    __shared__ int8_t lA[2][BM * LDS_STRIDE];
    __shared__ int8_t lB[2][BN * LDS_STRIDE];

    const int b     = blockIdx.z;
    const int mBase = blockIdx.y * BM;
    const int nBase = blockIdx.x * BN;

    const int8_t* Ab = A  + (size_t)b * MDIM * KDIM + (size_t)mBase * KDIM;
    const int8_t* Bb = Bm + (size_t)b * NDIM * KDIM + (size_t)nBase * KDIM;

    const float alpha = *alphap;

    const int tid   = threadIdx.x;
    const int lane  = tid & 31;
    const int wid   = tid >> 5;
    const int waveM = wid & 3;
    const int waveN = wid >> 2;
    const int half  = lane >> 4;
    const int r     = lane & 15;

    const int c0   = tid;
    const int c1   = tid + 256;
    const int row0 = c0 >> 2, off0 = (c0 & 3) * 16;
    const int row1 = c1 >> 2, off1 = (c1 & 3) * 16;

    v8i acc[2][4];
#pragma unroll
    for (int i = 0; i < 2; ++i)
#pragma unroll
        for (int j = 0; j < 4; ++j)
#pragma unroll
            for (int v = 0; v < 8; ++v)
                acc[i][j][v] = 0;

    auto compute_tile = [&](const int8_t* la, const int8_t* lb) {
        v8i afrag[2];
#pragma unroll
        for (int i = 0; i < 2; ++i) {
            const int8_t* p = la + (waveM * 32 + i * 16 + r) * LDS_STRIDE + half * 16;
            uint2 q0 = *(const uint2*)(p +  0);
            uint2 q1 = *(const uint2*)(p +  8);
            uint2 q2 = *(const uint2*)(p + 32);
            uint2 q3 = *(const uint2*)(p + 40);
            afrag[i][0] = (int)q0.x; afrag[i][1] = (int)q0.y;
            afrag[i][2] = (int)q1.x; afrag[i][3] = (int)q1.y;
            afrag[i][4] = (int)q2.x; afrag[i][5] = (int)q2.y;
            afrag[i][6] = (int)q3.x; afrag[i][7] = (int)q3.y;
        }
        v8i bfrag[4];
#pragma unroll
        for (int j = 0; j < 4; ++j) {
            const int8_t* p = lb + (waveN * 64 + j * 16 + r) * LDS_STRIDE + half * 16;
            uint4 q0 = *(const uint4*)(p +  0);
            uint4 q1 = *(const uint4*)(p + 32);
            bfrag[j][0] = (int)q0.x; bfrag[j][1] = (int)q0.y;
            bfrag[j][2] = (int)q0.z; bfrag[j][3] = (int)q0.w;
            bfrag[j][4] = (int)q1.x; bfrag[j][5] = (int)q1.y;
            bfrag[j][6] = (int)q1.z; bfrag[j][7] = (int)q1.w;
        }
#pragma unroll
        for (int i = 0; i < 2; ++i)
#pragma unroll
            for (int j = 0; j < 4; ++j)
                acc[i][j] = __builtin_amdgcn_wmma_i32_16x16x64_iu8(
                     true, afrag[i],
                     true, bfrag[j],
                    acc[i][j],
                     false,  false);
    };

#if HAVE_ASYNC_LDS
    auto issue_tile = [&](int k0, int8_t* la, int8_t* lb) {
        async_copy16(Ab + (size_t)row0 * KDIM + k0 + off0, la + row0 * LDS_STRIDE + off0);
        async_copy16(Ab + (size_t)row1 * KDIM + k0 + off1, la + row1 * LDS_STRIDE + off1);
        async_copy16(Bb + (size_t)row0 * KDIM + k0 + off0, lb + row0 * LDS_STRIDE + off0);
        async_copy16(Bb + (size_t)row1 * KDIM + k0 + off1, lb + row1 * LDS_STRIDE + off1);
    };

    int8_t* curA = &lA[0][0];
    int8_t* curB = &lB[0][0];
    int8_t* nxtA = &lA[1][0];
    int8_t* nxtB = &lB[1][0];

    issue_tile(0, curA, curB);
#pragma unroll 1
    for (int it = 0; it < KSTEPS - 1; ++it) {
        WAIT_ASYNC0();
        __syncthreads();
        issue_tile((it + 1) * BK, nxtA, nxtB);
        compute_tile(curA, curB);
        int8_t* t;
        t = curA; curA = nxtA; nxtA = t;
        t = curB; curB = nxtB; nxtB = t;
    }
    WAIT_ASYNC0();
    __syncthreads();
    compute_tile(curA, curB);
#else
#pragma unroll 1
    for (int k0 = 0; k0 < KDIM; k0 += BK) {
        uint4 ga0 = *(const uint4*)(Ab + (size_t)row0 * KDIM + k0 + off0);
        uint4 ga1 = *(const uint4*)(Ab + (size_t)row1 * KDIM + k0 + off1);
        uint4 gb0 = *(const uint4*)(Bb + (size_t)row0 * KDIM + k0 + off0);
        uint4 gb1 = *(const uint4*)(Bb + (size_t)row1 * KDIM + k0 + off1);
        __syncthreads();
        *(uint4*)(&lA[0][0] + row0 * LDS_STRIDE + off0) = ga0;
        *(uint4*)(&lA[0][0] + row1 * LDS_STRIDE + off1) = ga1;
        *(uint4*)(&lB[0][0] + row0 * LDS_STRIDE + off0) = gb0;
        *(uint4*)(&lB[0][0] + row1 * LDS_STRIDE + off1) = gb1;
        __syncthreads();
        compute_tile(&lA[0][0], &lB[0][0]);
    }
#endif

    __hip_bfloat16* outb = out + (size_t)b * MDIM * NDIM;
#pragma unroll
    for (int i = 0; i < 2; ++i) {
        const int mrow0 = mBase + waveM * 32 + i * 16 + half * 8;
#pragma unroll
        for (int j = 0; j < 4; ++j) {
            const int ncol = nBase + waveN * 64 + j * 16 + r;
#pragma unroll
            for (int v = 0; v < 8; ++v) {
                float f = alpha * (float)acc[i][j][v];
                __hip_bfloat16 h = __float2bfloat16(f);
                unsigned short u;
                __builtin_memcpy(&u, &h, 2);
                __builtin_nontemporal_store(
                    u, (unsigned short*)(outb + (size_t)(mrow0 + v) * NDIM + ncol));
            }
        }
    }
}

extern "C" void kernel_launch(void* const* d_in, const int* in_sizes, int n_in,
                              void* d_out, int out_size, void* d_ws, size_t ws_size,
                              hipStream_t stream) {
    (void)in_sizes; (void)n_in; (void)out_size; (void)d_ws; (void)ws_size;
    const int8_t* a      = (const int8_t*)d_in[0];
    const int8_t* bmat   = (const int8_t*)d_in[1];
    const float*  alphap = (const float*)d_in[2];
    __hip_bfloat16* out  = (__hip_bfloat16*)d_out;

    dim3 grid(NDIM / BN, MDIM / BM, BATCH);
    bmm_s8s8_bf16_wmma<<<grid, 256, 0, stream>>>(a, bmat, out, alphap);
}
